// Reducer_10754598109972
// MI455X (gfx1250) — matrix operand order repaired and checked by reading, not compiled, not run on hardware
//
#include <hip/hip_runtime.h>

// ---------- vector types ----------
typedef _Float16 h8   __attribute__((ext_vector_type(8)));
typedef _Float16 h16  __attribute__((ext_vector_type(16)));
typedef float    f8v  __attribute__((ext_vector_type(8)));
typedef float    f4v  __attribute__((ext_vector_type(4)));
typedef unsigned int u32x4 __attribute__((ext_vector_type(4)));
typedef int      i32x8 __attribute__((ext_vector_type(8)));
typedef int      i32x4 __attribute__((ext_vector_type(4)));

#if defined(__AMDGCN__) && defined(__has_builtin)
#if __has_builtin(__builtin_amdgcn_tensor_load_to_lds) && __has_builtin(__builtin_amdgcn_s_wait_tensorcnt)
#define USE_TDM 1
#endif
#endif
#ifndef USE_TDM
#define USE_TDM 0
#endif

// ---------- geometry ----------
#define COLS   784
#define KT1    25          // ceil(784/32) -> K padded to 800
#define NT1    8           // 128 cols / 16
#define NT2    4           // 64 / 16
#define KT2    4           // 128 / 32
#define NT3    2           // 32 / 16
#define KT3    2           // 64 / 32
#define FRAG_HALVES 512    // 32 lanes * 16 halves per B fragment tile
#define W1F_HALVES (KT1*NT1*FRAG_HALVES)   // 102400
#define W2F_HALVES (KT2*NT2*FRAG_HALVES)   // 8192
#define W3F_HALVES (KT3*NT3*FRAG_HALVES)   // 2048
#define TILE1_HALVES (NT1*FRAG_HALVES)     // 4096 halves = 8KB per k-tile of W1

// LDS layout (bytes):
//   h1  : [0, 34816)        128 rows x 136 halves   (phases 1-2)
//   Bbuf: [34816, 51200)    2 x 4096 halves (TDM dbl buffer, phase 1)
//   h2  : [34816, 53248)    128 rows x 72 halves    (phases 2-3, aliases Bbuf)
//   h3  : [0, 18432)        128 rows x 36 floats    (phase 3+, aliases h1)
#define SMEM_BYTES 53248
#define H1_STRIDE 136
#define H2_STRIDE 72
#define H3_STRIDE 36

// ---------- helpers ----------
__device__ __forceinline__ h16 cat16(h8 lo, h8 hi) {
  return __builtin_shufflevector(lo, hi, 0,1,2,3,4,5,6,7,8,9,10,11,12,13,14,15);
}

// 8 consecutive f32 -> 8 f16 (two b128 global loads + cvt)
__device__ __forceinline__ h8 cvt8(const float* p) {
  f4v x = *(const f4v*)p;
  f4v y = *(const f4v*)(p + 4);
  h8 r;
  r[0]=(_Float16)x[0]; r[1]=(_Float16)x[1]; r[2]=(_Float16)x[2]; r[3]=(_Float16)x[3];
  r[4]=(_Float16)y[0]; r[5]=(_Float16)y[1]; r[6]=(_Float16)y[2]; r[7]=(_Float16)y[3];
  return r;
}

// pre-swizzled B fragment: lane's 16 halves are contiguous
__device__ __forceinline__ h16 load_frag16(const _Float16* p) {
  h8 lo = *(const h8*)p;
  h8 hi = *(const h8*)(p + 8);
  return cat16(lo, hi);
}

// A fragment from an f16 LDS row (ISA 16-bit A layout: lane<16 holds K b..b+7 and b+16..b+23;
// lane>=16 holds b+8..b+15 and b+24..b+31 -> kbase already includes the +8 for hi half)
__device__ __forceinline__ h16 afrag_lds(const _Float16* rowp, int kbase) {
  h8 lo = *(const h8*)(rowp + kbase);
  h8 hi = *(const h8*)(rowp + kbase + 16);
  return cat16(lo, hi);
}

__device__ __forceinline__ f8v wmma_f16(h16 a, h16 b, f8v c) {
  return __builtin_amdgcn_wmma_f32_16x16x32_f16(false, a, false, b, (short)0, c, false, false);
}

#if USE_TDM
// 1-D TDM copy of 8192 bytes (1024 x 8B elements) global -> LDS
__device__ __forceinline__ void tdm_load_8k(unsigned lds_byte_addr, const void* gsrc) {
  unsigned long long ga = (unsigned long long)(size_t)gsrc;
  u32x4 g0;
  g0.x = 1u;                                            // D# count = 1, user mode
  g0.y = lds_byte_addr;                                 // lds_addr
  g0.z = (unsigned)ga;                                  // global_addr[31:0]
  g0.w = (unsigned)((ga >> 32) & 0x01FFFFFFu) | (2u << 30); // global_addr[56:32] | type=2
  i32x8 g1;
  g1[0] = (int)(3u << 16);                              // data_size = 8B
  g1[1] = (int)(1024u << 16);                           // tensor_dim0 = 1024 (lo16)
  g1[2] = 0;                                            // tensor_dim0 hi / tensor_dim1 lo
  g1[3] = (int)(1024u << 16);                           // tile_dim0 = 1024
  g1[4] = 0;                                            // tile_dim1/2 unused
  g1[5] = 1024;                                         // tensor_dim0_stride
  g1[6] = 0;
  g1[7] = 0;
  i32x4 gz4 = {0,0,0,0};
  i32x8 gz8 = {0,0,0,0,0,0,0,0};
  // 6-arg form (clang-23 / therock-10.0 headers): (g0, g1, g2, g3, g4, cpol)
  __builtin_amdgcn_tensor_load_to_lds(g0, g1, gz4, gz4, gz8, 0);
}
#endif

// ---------- kernel 1: repack weights into per-lane WMMA-B fragment order ----------
__device__ __forceinline__ void pack_one(int idx, const float* W, _Float16* out,
                                         int NT, int Kreal, int Ncols) {
  int e    = idx & 15;
  int lane = (idx >> 4) & 31;
  int nt   = (idx >> 9) % NT;
  int kt   = idx / (FRAG_HALVES * NT);
  int k = kt * 32 + ((lane & 16) ? 8 : 0) + (e < 8 ? e : e + 8);   // B layout: lane<16 -> K 0..7,16..23; lane>=16 -> K 8..15,24..31
  int n = nt * 16 + (lane & 15);
  float v = (k < Kreal) ? W[(size_t)k * Ncols + n] : 0.0f;
  out[idx] = (_Float16)v;
}

__global__ __launch_bounds__(256) void pack_weights(const float* W1, const float* W2, const float* W3,
                                                    _Float16* W1f, _Float16* W2f, _Float16* W3f) {
  int g = blockIdx.x * 256 + threadIdx.x;
  if (g < W1F_HALVES) {
    pack_one(g, W1, W1f, NT1, 784, 128);
  } else if (g < W1F_HALVES + W2F_HALVES) {
    pack_one(g - W1F_HALVES, W2, W2f, NT2, 128, 64);
  } else if (g < W1F_HALVES + W2F_HALVES + W3F_HALVES) {
    pack_one(g - W1F_HALVES - W2F_HALVES, W3, W3f, NT3, 64, 32);
  }
}

// ---------- kernel 2: fused MLP -> projected[N,2] ----------
__global__ __launch_bounds__(256) void mlp_kernel(const float* __restrict__ data,
                                                  const float* __restrict__ b1,
                                                  const float* __restrict__ b2,
                                                  const float* __restrict__ b3,
                                                  const float* __restrict__ Wout,
                                                  const float* __restrict__ bout,
                                                  const _Float16* __restrict__ W1f,
                                                  const _Float16* __restrict__ W2f,
                                                  const _Float16* __restrict__ W3f,
                                                  float* __restrict__ projected,
                                                  int N) {
  __shared__ __align__(32) unsigned char smem[SMEM_BYTES];
  _Float16* h1   = (_Float16*)smem;
  _Float16* Bbuf = (_Float16*)(smem + 34816);
  _Float16* h2   = (_Float16*)(smem + 34816);
  float*    h3   = (float*)smem;

  const int tid  = threadIdx.x;
  const int wave = tid >> 5;
  const int lane = tid & 31;
  const int l15  = lane & 15;
  const bool hiH = (lane & 16) != 0;
  const int koff = hiH ? 8 : 0;           // A layout hi-half K shift
  const int row0 = blockIdx.x * 128;

  int myrow = row0 + wave * 16 + l15;
  int arow  = myrow < N ? myrow : (N - 1);
  const float* rp = data + (size_t)arow * COLS;

  // ---- phase 1: h1 = relu(data @ W1 + b1), 16x128 per wave, K=784 (pad 800) ----
  f8v acc[NT1];
#pragma unroll
  for (int t = 0; t < NT1; ++t) acc[t] = (f8v){0,0,0,0,0,0,0,0};

#if USE_TDM
  unsigned lds0 = (unsigned)(size_t)(void*)Bbuf;
  unsigned lds1 = (unsigned)(size_t)(void*)(Bbuf + TILE1_HALVES);
  if (wave == 0) tdm_load_8k(lds0, W1f);               // prefetch k-tile 0
#endif

  for (int kt = 0; kt < KT1; ++kt) {
    __syncthreads();                                    // prev iter done reading buffers
#if USE_TDM
    if (wave == 0) {
      if (kt < KT1 - 1) {
        tdm_load_8k(((kt + 1) & 1) ? lds1 : lds0, W1f + (size_t)(kt + 1) * TILE1_HALVES);
        __builtin_amdgcn_s_wait_tensorcnt(1);           // oldest TDM (tile kt) landed
      } else {
        __builtin_amdgcn_s_wait_tensorcnt(0);
      }
    }
#else
    {
      unsigned* dst = (unsigned*)(Bbuf + (kt & 1) * TILE1_HALVES);
      const unsigned* src = (const unsigned*)(W1f + (size_t)kt * TILE1_HALVES);
      for (int t = tid; t < 2048; t += 256) dst[t] = src[t];
    }
#endif
    __syncthreads();                                    // tile kt visible to all waves

    // A fragment straight from global data (f32 -> f16)
    int kb = kt * 32 + koff;
    h8 z8 = {0,0,0,0,0,0,0,0};
    h8 c0 = (kb      < COLS) ? cvt8(rp + kb)      : z8;
    h8 c1 = (kb + 16 < COLS) ? cvt8(rp + kb + 16) : z8;
    h16 a = cat16(c0, c1);

    const _Float16* bb = Bbuf + (kt & 1) * TILE1_HALVES + lane * 16;
#pragma unroll
    for (int nt = 0; nt < NT1; ++nt) {
      h16 b = load_frag16(bb + nt * 32 * 16);
      acc[nt] = wmma_f16(a, b, acc[nt]);
    }
  }

  // bias + relu + f16 store to LDS (C layout: vgpr r, lanes<16 -> M=r, lanes>=16 -> M=r+8)
  const int srow0 = wave * 16 + (hiH ? 8 : 0);
#pragma unroll
  for (int nt = 0; nt < NT1; ++nt) {
    int col = nt * 16 + l15;
    float bias = b1[col];
#pragma unroll
    for (int r = 0; r < 8; ++r) {
      float v = acc[nt][r] + bias;
      v = v > 0.0f ? v : 0.0f;
      h1[(srow0 + r) * H1_STRIDE + col] = (_Float16)v;
    }
  }
  __syncthreads();   // h1 complete; Bbuf dead -> h2 region reusable

  // ---- phase 2: h2 = relu(h1 @ W2 + b2), K=128 ----
  f8v acc2[NT2];
#pragma unroll
  for (int t = 0; t < NT2; ++t) acc2[t] = (f8v){0,0,0,0,0,0,0,0};
  {
    const _Float16* h1row = h1 + (wave * 16 + l15) * H1_STRIDE;
#pragma unroll
    for (int kt = 0; kt < KT2; ++kt) {
      h16 a = afrag_lds(h1row, kt * 32 + koff);
      const _Float16* wf = W2f + (size_t)kt * NT2 * FRAG_HALVES + lane * 16;
#pragma unroll
      for (int nt = 0; nt < NT2; ++nt) {
        h16 b = load_frag16(wf + nt * 32 * 16);
        acc2[nt] = wmma_f16(a, b, acc2[nt]);
      }
    }
  }
#pragma unroll
  for (int nt = 0; nt < NT2; ++nt) {
    int col = nt * 16 + l15;
    float bias = b2[col];
#pragma unroll
    for (int r = 0; r < 8; ++r) {
      float v = acc2[nt][r] + bias;
      v = v > 0.0f ? v : 0.0f;
      h2[(srow0 + r) * H2_STRIDE + col] = (_Float16)v;
    }
  }
  __syncthreads();   // h2 complete; h1 dead -> h3 region reusable

  // ---- phase 3: h3 = relu(h2 @ W3 + b3), K=64 ----
  f8v acc3[NT3];
#pragma unroll
  for (int t = 0; t < NT3; ++t) acc3[t] = (f8v){0,0,0,0,0,0,0,0};
  {
    const _Float16* h2row = h2 + (wave * 16 + l15) * H2_STRIDE;
#pragma unroll
    for (int kt = 0; kt < KT3; ++kt) {
      h16 a = afrag_lds(h2row, kt * 32 + koff);
      const _Float16* wf = W3f + (size_t)kt * NT3 * FRAG_HALVES + lane * 16;
#pragma unroll
      for (int nt = 0; nt < NT3; ++nt) {
        h16 b = load_frag16(wf + nt * 32 * 16);
        acc3[nt] = wmma_f16(a, b, acc3[nt]);
      }
    }
  }
#pragma unroll
  for (int nt = 0; nt < NT3; ++nt) {
    int col = nt * 16 + l15;
    float bias = b3[col];
#pragma unroll
    for (int r = 0; r < 8; ++r) {
      float v = acc3[nt][r] + bias;
      v = v > 0.0f ? v : 0.0f;
      h3[(srow0 + r) * H3_STRIDE + col] = v;
    }
  }
  __syncthreads();   // h3 complete

  // ---- final: projected = h3 @ Wout + bout (OUT=2), one (row,out) per thread ----
  {
    int prow = tid >> 1;
    int o    = tid & 1;
    int grow = row0 + prow;
    if (grow < N) {
      const float* hrow = h3 + prow * H3_STRIDE;
      float s = bout[o];
#pragma unroll
      for (int c = 0; c < 32; ++c) s += hrow[c] * Wout[c * 2 + o];
      projected[(size_t)grow * 2 + o] = s;
    }
  }
}

// ---------- kernel 3: gather + squared distance ----------
__global__ __launch_bounds__(256) void dist_kernel(const float* __restrict__ projected,
                                                   const int* __restrict__ idxs,
                                                   float* __restrict__ out,
                                                   int total, int K) {
  int g = blockIdx.x * 256 + threadIdx.x;
  if (g >= total) return;
  int i  = g / K;
  int nb = idxs[g];
  const float2* P = (const float2*)projected;
  float2 a = P[i];
  float2 b = P[nb];
  float dx = a.x - b.x;
  float dy = a.y - b.y;
  out[g] = dx * dx + dy * dy;
}

// ---------- host launcher ----------
extern "C" void kernel_launch(void* const* d_in, const int* in_sizes, int n_in,
                              void* d_out, int out_size, void* d_ws, size_t ws_size,
                              hipStream_t stream) {
  const float* data = (const float*)d_in[0];
  const int*   idxs = (const int*)d_in[1];
  const float* W1   = (const float*)d_in[2];
  const float* b1   = (const float*)d_in[3];
  const float* W2   = (const float*)d_in[4];
  const float* b2   = (const float*)d_in[5];
  const float* W3   = (const float*)d_in[6];
  const float* b3   = (const float*)d_in[7];
  const float* Wout = (const float*)d_in[8];
  const float* bout = (const float*)d_in[9];

  const int N = in_sizes[0] / COLS;        // 200000
  const int K = in_sizes[1] / N;           // 10
  const int total = N * K;

  char* ws = (char*)d_ws;
  _Float16* W1f = (_Float16*)(ws);                       // 204800 B
  _Float16* W2f = (_Float16*)(ws + 204800);              // 16384 B
  _Float16* W3f = (_Float16*)(ws + 221184);              // 4096 B
  float* projected = (float*)(ws + 225280);              // N*2*4 B

  const int packN = W1F_HALVES + W2F_HALVES + W3F_HALVES;
  pack_weights<<<(packN + 255) / 256, 256, 0, stream>>>(W1, W2, W3, W1f, W2f, W3f);

  const int blocks = (N + 127) / 128;
  mlp_kernel<<<blocks, 256, 0, stream>>>(data, b1, b2, b3, Wout, bout,
                                         W1f, W2f, W3f, projected, N);

  dist_kernel<<<(total + 255) / 256, 256, 0, stream>>>(projected, idxs, (float*)d_out, total, K);
}
